// ana_block_33560874451099
// MI455X (gfx1250) — matrix operand order repaired and checked by reading, not compiled, not run on hardware
//
#include <hip/hip_runtime.h>
#include <hip/hip_bf16.h>
#include <math.h>

// Problem constants (from reference)
#define Bb   8
#define Cc   256
#define Nn   2048
#define Hh   4
#define Dd   64
#define BHN  (Bb * Hh * Nn)   // 65536
#define BNp  (Bb * Nn)        // 16384

// Input index map (jax pytree flatten: dict keys sorted; top level {params,x}):
//  0 bk 1 bq 2 cb0 3 cb1 4 cbn_b 5 cbn_g 6 cw0 7 cw1 8 fcb 9 fcw 10 ln_b 11 ln_g
// 12-15 nb[32,64,128,256] 16-18 nbn_b 19-21 nbn_g 22 nln_b 23 nln_g
// 24-27 nw[32x4,64x32,128x64,256x128] 28 wk 29 wq 30 zoom 31 x

typedef __attribute__((ext_vector_type(16))) __bf16 v16bf;
typedef __attribute__((ext_vector_type(8)))  __bf16 v8bf;
typedef __attribute__((ext_vector_type(4)))  __bf16 v4bf;
typedef __attribute__((ext_vector_type(8)))  float  v8f;
typedef __attribute__((ext_vector_type(4)))  float  f4;

// Build a 16-element A/B fragment from two contiguous 8x bf16 chunks (ISA 7.12.2
// striping: each lane owns two 8-wide contiguous K runs).
static __device__ __forceinline__ v16bf frag16(const __bf16* p0, const __bf16* p1) {
  v8bf lo = *(const v8bf*)p0;
  v8bf hi = *(const v8bf*)p1;
  v16bf r;
#pragma unroll
  for (int i = 0; i < 8; ++i) { r[i] = lo[i]; r[i + 8] = hi[i]; }
  return r;
}

static __device__ __forceinline__ v8f wmma_bf16(v16bf a, v16bf b, v8f c) {
  return __builtin_amdgcn_wmma_f32_16x16x32_bf16(false, a, false, b, (short)0, c,
                                                 false, false);
}

// CDNA5 async global->LDS copy (ASYNCcnt-tracked, no VGPR round trip).
// LDS operand is the byte offset within the wave's LDS allocation == low 32
// bits of the generic pointer to a __shared__ object (flat aperture, ISA 10.2).
static __device__ __forceinline__ void async_g2l_b128(void* lds, const void* g) {
  unsigned           ldsoff = (unsigned)(size_t)lds;
  unsigned long long gaddr  = (unsigned long long)(size_t)g;
  asm volatile("global_load_async_to_lds_b128 %0, %1, off"
               :: "v"(ldsoff), "v"(gaddr) : "memory");
}
static __device__ __forceinline__ void wait_async0() {
  asm volatile("s_wait_asynccnt 0" ::: "memory");
}

// ---------------------------------------------------------------------------
// Generic 1x1-conv GEMM:  Y[b] = W (Cout x Cin) * X[b] (Cin x N) + bias
// 128x128 output tile per workgroup, 8 waves = 4x2 wave grid, each wave 32x64
// (2x4 WMMA tiles). bf16 multiply, f32 accumulate.
// mode 0: Y f32 [B][Cout][N].  mode 1: bf16 head-major [B][H][N][DIM] (c=d*H+h).
// ---------------------------------------------------------------------------
__global__ __launch_bounds__(256) void gemm_bf16_k(
    const float* __restrict__ W, const float* __restrict__ bias,
    const float* __restrict__ X, float* __restrict__ Y,
    __bf16* __restrict__ Yqk, int Cout, int Cin, int mode) {
  __shared__ __bf16 As[128][32];  // [cout_local][k]
  __shared__ __bf16 Bs[128][32];  // [n_local][k]  (X staged transposed)
  const int b    = blockIdx.z;
  const int row0 = blockIdx.y * 128;
  const int col0 = blockIdx.x * 128;
  const float* Xb = X + (size_t)b * Cin * Nn;
  const int tid = threadIdx.x, wid = tid >> 5, lane = tid & 31;
  const int half = lane >> 4, l16 = lane & 15, kb = half * 8;
  const int wrow = (wid & 3) * 32, wcol = (wid >> 2) * 64;

  v8f acc[2][4] = {};

  for (int k0 = 0; k0 < Cin; k0 += 32) {
    __syncthreads();
    // A tile: b128 loads, packed bf16 b64 stores (4 f32 -> 4 bf16 per step)
    for (int i = tid; i < 128 * 8; i += 256) {
      int r = i >> 3, c = (i & 7) * 4;
      f4 w = *(const f4*)&W[(size_t)(row0 + r) * Cin + (k0 + c)];
      v4bf o;
#pragma unroll
      for (int j = 0; j < 4; j++) o[j] = (__bf16)w[j];
      *(v4bf*)&As[r][c] = o;
    }
    // B tile: b128 loads coalesced over n, transposed bf16 scatter into LDS
    for (int i = tid; i < 32 * 32; i += 256) {
      int k = i >> 5, n = (i & 31) * 4;
      f4 xv = *(const f4*)&Xb[(size_t)(k0 + k) * Nn + (col0 + n)];
#pragma unroll
      for (int j = 0; j < 4; j++) Bs[n + j][k] = (__bf16)xv[j];
    }
    if (k0 + 32 < Cin) {  // global_prefetch_b8 of next k-tiles
      __builtin_prefetch(&W[(size_t)(row0 + (tid >> 1)) * Cin + k0 + 32], 0, 1);
      __builtin_prefetch(&Xb[(size_t)(k0 + 32 + (tid >> 3)) * Nn + col0], 0, 1);
    }
    __syncthreads();

    v16bf af[2], bf[4];
#pragma unroll
    for (int rt = 0; rt < 2; rt++) {
      const __bf16* ap = &As[wrow + rt * 16 + l16][0];
      af[rt] = frag16(ap + kb, ap + kb + 16);
    }
#pragma unroll
    for (int ct = 0; ct < 4; ct++) {
      const __bf16* bp = &Bs[wcol + ct * 16 + l16][kb];
      bf[ct] = frag16(bp, bp + 16);
    }
#pragma unroll
    for (int rt = 0; rt < 2; rt++)
#pragma unroll
      for (int ct = 0; ct < 4; ct++)
        acc[rt][ct] = wmma_bf16(af[rt], bf[ct], acc[rt][ct]);
  }

#pragma unroll
  for (int rt = 0; rt < 2; rt++)
#pragma unroll
    for (int ct = 0; ct < 4; ct++)
#pragma unroll
      for (int r = 0; r < 8; r++) {
        int m = row0 + wrow + rt * 16 + r + 8 * half;  // C/D striping
        int n = col0 + wcol + ct * 16 + l16;
        float v = acc[rt][ct][r] + bias[m];
        if (mode == 0) {
          Y[((size_t)b * Cout + m) * Nn + n] = v;
        } else {
          int d = m >> 2, h = m & 3;  // c = d*H + h
          Yqk[(((size_t)b * Hh + h) * Nn + n) * Dd + d] = (__bf16)v;
        }
      }
}

// ---------------------------------------------------------------------------
// Flash attention pass 1: per (b,h), row max and sum(exp) for each query.
// Q block (contiguous 128x64 bf16) staged with async global->LDS b128 DMA.
// ---------------------------------------------------------------------------
__global__ __launch_bounds__(256) void attn_pass1_k(
    const __bf16* __restrict__ Qb, const __bf16* __restrict__ Kb,
    float* __restrict__ rowmax, float* __restrict__ rowsum) {
  __shared__ __bf16 Qs[128][64];
  const int bh = blockIdx.y;
  const int n0 = blockIdx.x * 128;
  const __bf16* Q = Qb + (size_t)bh * Nn * Dd;
  const __bf16* K = Kb + (size_t)bh * Nn * Dd;
  const int tid = threadIdx.x, wid = tid >> 5, lane = tid & 31;
  const int half = lane >> 4, l16 = lane & 15, kb = half * 8;

  {
    const __bf16* src = Q + (size_t)n0 * Dd;   // contiguous 16KB block
    __bf16* dst = &Qs[0][0];
    for (int i = tid; i < 1024; i += 256)      // 1024 x b128 chunks
      async_g2l_b128(dst + i * 8, src + i * 8);
    wait_async0();
  }
  __syncthreads();

  const int qrow = wid * 16;
  v16bf aq[2];
#pragma unroll
  for (int dk = 0; dk < 2; dk++) {
    const __bf16* ap = &Qs[qrow + l16][dk * 32];
    aq[dk] = frag16(ap + kb, ap + kb + 16);
  }

  float rmax[8], rsum[8];
#pragma unroll
  for (int r = 0; r < 8; r++) { rmax[r] = -3.0e38f; rsum[r] = 0.0f; }
  const float scale = 0.125f;  // 1/sqrt(64)

  for (int m0 = 0; m0 < Nn; m0 += 16) {
    v8f s = {};
#pragma unroll
    for (int dk = 0; dk < 2; dk++) {  // B-frag straight from K[m][d] rows
      const __bf16* bp = &K[(size_t)(m0 + l16) * Dd + dk * 32 + kb];
      s = wmma_bf16(aq[dk], frag16(bp, bp + 16), s);
    }
#pragma unroll
    for (int r = 0; r < 8; r++) {  // online softmax, 16-lane row reductions
      float v = s[r] * scale;
      float mx = v;
#pragma unroll
      for (int off = 1; off < 16; off <<= 1) mx = fmaxf(mx, __shfl_xor(mx, off, 32));
      float e = __expf(v - mx);
#pragma unroll
      for (int off = 1; off < 16; off <<= 1) e += __shfl_xor(e, off, 32);
      float nm = fmaxf(rmax[r], mx);
      rsum[r] = rsum[r] * __expf(rmax[r] - nm) + e * __expf(mx - nm);
      rmax[r] = nm;
    }
  }
  if (l16 == 0) {
#pragma unroll
    for (int r = 0; r < 8; r++) {
      int row = n0 + qrow + r + 8 * half;
      rowmax[(size_t)bh * Nn + row] = rmax[r];
      rowsum[(size_t)bh * Nn + row] = rsum[r];
    }
  }
}

// ---------------------------------------------------------------------------
// Flash attention pass 2: exact P (stats from pass1), O = P*V (V==K), plus
// column sums of P and P^2 (h_star) via LDS ds_add_f32 then global atomics.
// ---------------------------------------------------------------------------
__global__ __launch_bounds__(256) void attn_pass2_k(
    const __bf16* __restrict__ Qb, const __bf16* __restrict__ Kb,
    const float* __restrict__ rowmax, const float* __restrict__ rowsum,
    float* __restrict__ attn, float* __restrict__ hs1, float* __restrict__ hs2) {
  __shared__ __bf16 Qs[128][64];
  __shared__ __bf16 Vt[64][32];        // V chunk transposed: [d][m_local]
  __shared__ __bf16 Pb[8][16][32];     // per-wave P tile for A-frag reload
  __shared__ float  cs1[Nn];
  __shared__ float  cs2[Nn];
  const int bh = blockIdx.y, b = bh >> 2, h = bh & 3;
  const int n0 = blockIdx.x * 128;
  const __bf16* Q = Qb + (size_t)bh * Nn * Dd;
  const __bf16* K = Kb + (size_t)bh * Nn * Dd;
  const int tid = threadIdx.x, wid = tid >> 5, lane = tid & 31;
  const int half = lane >> 4, l16 = lane & 15, kb = half * 8;

  {
    const __bf16* src = Q + (size_t)n0 * Dd;
    __bf16* dst = &Qs[0][0];
    for (int i = tid; i < 1024; i += 256)
      async_g2l_b128(dst + i * 8, src + i * 8);
  }
  for (int i = tid; i < Nn; i += 256) { cs1[i] = 0.0f; cs2[i] = 0.0f; }
  wait_async0();
  __syncthreads();

  const int qrow = wid * 16;
  v16bf aq[2];
#pragma unroll
  for (int dk = 0; dk < 2; dk++) {
    const __bf16* ap = &Qs[qrow + l16][dk * 32];
    aq[dk] = frag16(ap + kb, ap + kb + 16);
  }
  float rm[8], ri[8];
#pragma unroll
  for (int r = 0; r < 8; r++) {
    int row = n0 + qrow + r + 8 * half;
    rm[r] = rowmax[(size_t)bh * Nn + row];
    ri[r] = 1.0f / rowsum[(size_t)bh * Nn + row];
  }
  v8f po[4] = {};
  const float scale = 0.125f;

  for (int m0 = 0; m0 < Nn; m0 += 32) {
    __syncthreads();
    // stage V chunk transposed: b128 loads along d, bf16 scatter
    for (int i = tid; i < 32 * 8; i += 256) {
      int ml = i >> 3, d0 = (i & 7) * 8;
      v8bf kv = *(const v8bf*)&K[(size_t)(m0 + ml) * Dd + d0];
#pragma unroll
      for (int j = 0; j < 8; j++) Vt[d0 + j][ml] = kv[j];
    }
    __syncthreads();

#pragma unroll
    for (int ct = 0; ct < 2; ct++) {  // two 16-col score tiles -> 32-wide P
      v8f s = {};
#pragma unroll
      for (int dk = 0; dk < 2; dk++) {
        const __bf16* bp = &K[(size_t)(m0 + ct * 16 + l16) * Dd + dk * 32 + kb];
        s = wmma_bf16(aq[dk], frag16(bp, bp + 16), s);
      }
      float ps = 0.0f, ps2 = 0.0f;
#pragma unroll
      for (int r = 0; r < 8; r++) {
        float p = __expf(s[r] * scale - rm[r]) * ri[r];
        Pb[wid][r + 8 * half][ct * 16 + l16] = (__bf16)p;
        ps += p; ps2 += p * p;
      }
      ps  += __shfl_xor(ps, 16, 32);   // fold rows 0-7 with 8-15 (same column)
      ps2 += __shfl_xor(ps2, 16, 32);
      if (half == 0) {                  // ds_add_f32 column accumulators
        atomicAdd(&cs1[m0 + ct * 16 + l16], ps);
        atomicAdd(&cs2[m0 + ct * 16 + l16], ps2);
      }
    }
    __syncthreads();

    const __bf16* pp = &Pb[wid][l16][0];   // P as A-frag (16x32, K = keys)
    v16bf pf = frag16(pp + kb, pp + kb + 16);
#pragma unroll
    for (int dt = 0; dt < 4; dt++) {       // O += P * V over 4 d-tiles
      const __bf16* vp = &Vt[dt * 16 + l16][kb];
      po[dt] = wmma_bf16(pf, frag16(vp, vp + 16), po[dt]);
    }
  }

#pragma unroll
  for (int dt = 0; dt < 4; dt++)
#pragma unroll
    for (int r = 0; r < 8; r++) {
      int d = dt * 16 + l16;
      int n = n0 + qrow + r + 8 * half;
      attn[((size_t)b * Cc + d * Hh + h) * Nn + n] = po[dt][r];
    }
  __syncthreads();
  for (int i = tid; i < Nn; i += 256) {  // one global atomic per column
    atomicAdd(&hs1[(size_t)bh * Nn + i], cs1[i]);
    atomicAdd(&hs2[(size_t)bh * Nn + i], cs2[i]);
  }
}

// --------------------------- small tail kernels ----------------------------
__global__ __launch_bounds__(256) void zero_k(float* __restrict__ p, int n) {
  int i = blockIdx.x * 256 + threadIdx.x;
  if (i < n) p[i] = 0.0f;
}

__global__ __launch_bounds__(256) void confidence_k(
    const float* __restrict__ hs1, const float* __restrict__ hs2,
    const float* __restrict__ zoom, float* __restrict__ conf) {
  int i = blockIdx.x * 256 + threadIdx.x;
  if (i >= BHN) return;
  int h = (i / Nn) & 3;
  float v = zoom[h] * (hs1[i] - hs2[i]);
  conf[i] = 1.0f / (1.0f + __expf(-v));
}

__global__ __launch_bounds__(256) void small_conv_k(
    const float* __restrict__ W, const float* __restrict__ bias,
    const float* __restrict__ X, float* __restrict__ Y, int Cout, int Cin) {
  int idx = blockIdx.x * 256 + threadIdx.x;
  int total = Bb * Cout * Nn;
  if (idx >= total) return;
  int n = idx % Nn;
  int co = (idx / Nn) % Cout;
  int b = idx / (Nn * Cout);
  const float* xb = X + (size_t)b * Cin * Nn + n;
  const float* wr = W + (size_t)co * Cin;  // uniform per wave -> scalar loads
  float acc = bias[co];
  for (int ci = 0; ci < Cin; ci++) acc += wr[ci] * xb[(size_t)ci * Nn];
  Y[idx] = acc;
}

// Train-mode BatchNorm over (batch, length) per channel, fused ReLU, in place.
__global__ __launch_bounds__(256) void bn_relu_k(
    float* __restrict__ X, const float* __restrict__ g,
    const float* __restrict__ bta, int C) {
  __shared__ float s1[256], s2[256];
  const int c = blockIdx.x, tid = threadIdx.x;
  float a = 0.0f, q = 0.0f;
  for (int i = tid; i < BNp; i += 256) {
    int b = i >> 11, n = i & (Nn - 1);
    float v = X[((size_t)b * C + c) * Nn + n];
    a += v; q += v * v;
  }
  s1[tid] = a; s2[tid] = q;
  __syncthreads();
  for (int s = 128; s > 0; s >>= 1) {
    if (tid < s) { s1[tid] += s1[tid + s]; s2[tid] += s2[tid + s]; }
    __syncthreads();
  }
  float mean = s1[0] * (1.0f / BNp);
  float var  = s2[0] * (1.0f / BNp) - mean * mean;
  float inv  = rsqrtf(var + 1e-5f);
  float gg = g[c], bb = bta[c];
  for (int i = tid; i < BNp; i += 256) {
    int b = i >> 11, n = i & (Nn - 1);
    size_t off = ((size_t)b * C + c) * Nn + n;
    X[off] = fmaxf((X[off] - mean) * inv * gg + bb, 0.0f);
  }
}

// LayerNorm over channels at each (b,n); optional residual add of xin.
__global__ __launch_bounds__(256) void ln_k(
    const float* __restrict__ Xn, const float* __restrict__ g,
    const float* __restrict__ bta, const float* __restrict__ xin,
    float* __restrict__ out) {
  __shared__ float s1[256], s2[256];
  int pos = blockIdx.x;
  int b = pos >> 11, n = pos & (Nn - 1);
  int c = threadIdx.x;
  size_t off = ((size_t)b * Cc + c) * Nn + n;
  float v = Xn[off];
  s1[c] = v; s2[c] = v * v;
  __syncthreads();
  for (int s = 128; s > 0; s >>= 1) {
    if (c < s) { s1[c] += s1[c + s]; s2[c] += s2[c + s]; }
    __syncthreads();
  }
  float mean = s1[0] * (1.0f / Cc);
  float var  = s2[0] * (1.0f / Cc) - mean * mean;
  float r = (v - mean) * rsqrtf(var + 1e-5f) * g[c] + bta[c];
  if (xin) r += xin[off];
  out[off] = r;
}

__global__ __launch_bounds__(256) void concat_k(
    const float* __restrict__ A, const float* __restrict__ Bf,
    float* __restrict__ out) {
  int idx = blockIdx.x * 256 + threadIdx.x;
  int total = Bb * 2 * Cc * Nn;
  if (idx >= total) return;
  int n = idx % Nn;
  int c = (idx / Nn) % (2 * Cc);
  int b = idx / (Nn * 2 * Cc);
  out[idx] = (c < Cc) ? A[((size_t)b * Cc + c) * Nn + n]
                      : Bf[((size_t)b * Cc + (c - Cc)) * Nn + n];
}

// ---------------------------------------------------------------------------
extern "C" void kernel_launch(void* const* d_in, const int* in_sizes, int n_in,
                              void* d_out, int out_size, void* d_ws, size_t ws_size,
                              hipStream_t stream) {
  (void)in_sizes; (void)n_in; (void)out_size; (void)ws_size;
  const float* bk    = (const float*)d_in[0];
  const float* bq    = (const float*)d_in[1];
  const float* cb0   = (const float*)d_in[2];
  const float* cb1   = (const float*)d_in[3];
  const float* cbn_b = (const float*)d_in[4];
  const float* cbn_g = (const float*)d_in[5];
  const float* cw0   = (const float*)d_in[6];
  const float* cw1   = (const float*)d_in[7];
  const float* fcb   = (const float*)d_in[8];
  const float* fcw   = (const float*)d_in[9];
  const float* ln_b  = (const float*)d_in[10];
  const float* ln_g  = (const float*)d_in[11];
  const float* nb0   = (const float*)d_in[12];
  const float* nb1   = (const float*)d_in[13];
  const float* nb2   = (const float*)d_in[14];
  const float* nb3   = (const float*)d_in[15];
  const float* nbnb0 = (const float*)d_in[16];
  const float* nbnb1 = (const float*)d_in[17];
  const float* nbnb2 = (const float*)d_in[18];
  const float* nbng0 = (const float*)d_in[19];
  const float* nbng1 = (const float*)d_in[20];
  const float* nbng2 = (const float*)d_in[21];
  const float* nln_b = (const float*)d_in[22];
  const float* nln_g = (const float*)d_in[23];
  const float* nw0   = (const float*)d_in[24];
  const float* nw1   = (const float*)d_in[25];
  const float* nw2   = (const float*)d_in[26];
  const float* nw3   = (const float*)d_in[27];
  const float* wk    = (const float*)d_in[28];
  const float* wq    = (const float*)d_in[29];
  const float* zoom  = (const float*)d_in[30];
  const float* x     = (const float*)d_in[31];
  float* out = (float*)d_out;

  size_t off = 0;
  auto carve = [&](size_t bytes) -> void* {
    void* p = (char*)d_ws + off;
    off += (bytes + 255) & ~(size_t)255;
    return p;
  };
  __bf16* Qb    = (__bf16*)carve((size_t)BHN * Dd * 2);
  __bf16* Kbuf  = (__bf16*)carve((size_t)BHN * Dd * 2);
  float* rowmax = (float*)carve((size_t)BHN * 4);
  float* rowsum = (float*)carve((size_t)BHN * 4);
  float* hs1    = (float*)carve((size_t)BHN * 4);
  float* hs2    = (float*)carve((size_t)BHN * 4);
  float* attn   = (float*)carve((size_t)Bb * Cc * Nn * 4);
  float* cfeat  = (float*)carve((size_t)Bb * Cc * Nn * 4);
  float* conf   = (float*)carve((size_t)BHN * 4);
  float* mbuf0  = (float*)carve((size_t)Bb * 32 * Nn * 4);
  float* mbuf1  = (float*)carve((size_t)Bb * 64 * Nn * 4);
  float* mbuf2  = (float*)carve((size_t)Bb * 128 * Nn * 4);
  float* mbuf3  = (float*)carve((size_t)Bb * Cc * Nn * 4);
  float* xpn    = (float*)carve((size_t)Bb * Cc * Nn * 4);
  float* fcat   = (float*)carve((size_t)Bb * 2 * Cc * Nn * 4);
  float* y0     = (float*)carve((size_t)Bb * Cc * Nn * 4);
  float* f2     = (float*)carve((size_t)Bb * Cc * Nn * 4);

  dim3 blk(256);
  // zero h_star accumulators
  zero_k<<<(BHN + 255) / 256, blk, 0, stream>>>(hs1, BHN);
  zero_k<<<(BHN + 255) / 256, blk, 0, stream>>>(hs2, BHN);

  // Q/K projections -> bf16 head-major  (v==k per reference)
  dim3 gproj(Nn / 128, Cc / 128, Bb);
  gemm_bf16_k<<<gproj, blk, 0, stream>>>(wq, bq, x, nullptr, Qb, Cc, Cc, 1);
  gemm_bf16_k<<<gproj, blk, 0, stream>>>(wk, bk, x, nullptr, Kbuf, Cc, Cc, 1);

  // flash attention
  dim3 gatt(Nn / 128, Bb * Hh);
  attn_pass1_k<<<gatt, blk, 0, stream>>>(Qb, Kbuf, rowmax, rowsum);
  attn_pass2_k<<<gatt, blk, 0, stream>>>(Qb, Kbuf, rowmax, rowsum, attn, hs1, hs2);

  // c_feat = fcw * attn + fcb
  gemm_bf16_k<<<gproj, blk, 0, stream>>>(fcw, fcb, attn, cfeat, nullptr, Cc, Cc, 0);

  // confidence branch
  confidence_k<<<(BHN + 255) / 256, blk, 0, stream>>>(hs1, hs2, zoom, conf);
  small_conv_k<<<(Bb * 32 * Nn + 255) / 256, blk, 0, stream>>>(nw0, nb0, conf, mbuf0, 32, Hh);
  bn_relu_k<<<32, blk, 0, stream>>>(mbuf0, nbng0, nbnb0, 32);
  small_conv_k<<<(Bb * 64 * Nn + 255) / 256, blk, 0, stream>>>(nw1, nb1, mbuf0, mbuf1, 64, 32);
  bn_relu_k<<<64, blk, 0, stream>>>(mbuf1, nbng1, nbnb1, 64);
  small_conv_k<<<(Bb * 128 * Nn + 255) / 256, blk, 0, stream>>>(nw2, nb2, mbuf1, mbuf2, 128, 64);
  bn_relu_k<<<128, blk, 0, stream>>>(mbuf2, nbng2, nbnb2, 128);
  small_conv_k<<<(Bb * Cc * Nn + 255) / 256, blk, 0, stream>>>(nw3, nb3, mbuf2, mbuf3, Cc, 128);
  // layernorm + residual add:  xpn = x + LN(mbuf3)
  ln_k<<<BNp, blk, 0, stream>>>(mbuf3, nln_g, nln_b, x, xpn);

  // merge: concat -> conv(2C->C) -> BN+ReLU -> conv(C->C) -> LN
  concat_k<<<(Bb * 2 * Cc * Nn + 255) / 256, blk, 0, stream>>>(xpn, cfeat, fcat);
  gemm_bf16_k<<<gproj, blk, 0, stream>>>(cw0, cb0, fcat, y0, nullptr, Cc, 2 * Cc, 0);
  bn_relu_k<<<Cc, blk, 0, stream>>>(y0, cbn_g, cbn_b, Cc);
  gemm_bf16_k<<<gproj, blk, 0, stream>>>(cw1, cb1, y0, f2, nullptr, Cc, Cc, 0);
  ln_k<<<BNp, blk, 0, stream>>>(f2, ln_g, ln_b, nullptr, out);
}
